// Attention_22179211116942
// MI455X (gfx1250) — matrix operand order repaired and checked by reading, not compiled, not run on hardware
//
#include <hip/hip_runtime.h>
#include <hip/hip_bf16.h>

// ---------------------------------------------------------------------------
// Types for CDNA5 WMMA (wave32): V_WMMA_F32_16X16X32_BF16
// ---------------------------------------------------------------------------
typedef __attribute__((ext_vector_type(16))) __bf16 v16bf;
typedef __attribute__((ext_vector_type(8)))  __bf16 v8bf;
typedef __attribute__((ext_vector_type(8)))  float  v8f;

__device__ __forceinline__ v8f wmma_bf16(v16bf a, v16bf b, v8f c) {
    // (neg_a, A, neg_b, B, c_mod, C, reuse_a, reuse_b)
    return __builtin_amdgcn_wmma_f32_16x16x32_bf16(false, a, false, b,
                                                   (short)0, c, false, false);
}

__device__ __forceinline__ v8f zero_v8f() {
    v8f z;
#pragma unroll
    for (int i = 0; i < 8; ++i) z[i] = 0.0f;
    return z;
}

// A fragment (16x32 bf16, M x K), loading from row-major f32 with convert.
// Lane layout: lane row M=lane&15 holds K=16h..16h+15
// with h = lane>>4.
__device__ __forceinline__ v16bf load_a_f32(const float* __restrict__ base,
                                            int ld, int row0, int k0,
                                            int lm, int half) {
    const float* r = base + (size_t)(row0 + lm) * ld + k0;
    v16bf a;
#pragma unroll
    for (int j = 0; j < 8; ++j) {
        a[j]     = (__bf16)r[16 * half + j];
        a[8 + j] = (__bf16)r[16 * half + 8 + j];
    }
    return a;
}

// A fragment from row-major bf16 (vector loads).
__device__ __forceinline__ v16bf load_a_bf16(const __bf16* __restrict__ base,
                                             int ld, int row0, int k0,
                                             int lm, int half) {
    const __bf16* r = base + (size_t)(row0 + lm) * ld + k0;
    v8bf lo = *reinterpret_cast<const v8bf*>(r + 16 * half);
    v8bf hi = *reinterpret_cast<const v8bf*>(r + 16 * half + 8);
    v16bf a;
#pragma unroll
    for (int j = 0; j < 8; ++j) { a[j] = lo[j]; a[8 + j] = hi[j]; }
    return a;
}

// B fragment (32x16 bf16, K x N) where B[kk][n] = W[o0+n][i0+kk], W row-major
// f32 (torch Linear: y = x @ W^T). Lane n=lane&15 holds K = 16*half + 0..15,
// which is 16 contiguous f32 along W's row -> convert.
__device__ __forceinline__ v16bf load_bT_f32(const float* __restrict__ w,
                                             int ld, int o0, int i0,
                                             int lm, int half) {
    const float* r = w + (size_t)(o0 + lm) * ld + i0 + 16 * half;
    v16bf b;
#pragma unroll
    for (int j = 0; j < 16; ++j) b[j] = (__bf16)r[j];
    return b;
}

// B fragment where each column n is 32 contiguous bf16 at colbase(n); lane
// takes elements [16*half .. 16*half+15].
__device__ __forceinline__ v16bf load_b_contig_bf16(const __bf16* __restrict__ col,
                                                    int half) {
    v8bf lo = *reinterpret_cast<const v8bf*>(col + 16 * half);
    v8bf hi = *reinterpret_cast<const v8bf*>(col + 16 * half + 8);
    v16bf b;
#pragma unroll
    for (int j = 0; j < 8; ++j) { b[j] = lo[j]; b[8 + j] = hi[j]; }
    return b;
}

// ---------------------------------------------------------------------------
// Kernel 1: fused Q/K/V/G projections.  Rows = flattened (s,q) = 65536.
// grid.x = row tile (16 rows), grid.y = projection id {0:Q,1:K,2:V,3:G}.
// 256 threads = 8 waves; wave w computes output columns [16w, 16w+16).
// ---------------------------------------------------------------------------
__global__ __launch_bounds__(256) void proj_kernel(
    const float* __restrict__ qx, const float* __restrict__ kvx,
    const float* __restrict__ wq, const float* __restrict__ wk,
    const float* __restrict__ wv, const float* __restrict__ wg,
    const float* __restrict__ bg,
    __bf16* __restrict__ qb,   // [s][h][q][32]
    __bf16* __restrict__ kb,   // [s][h][k][32]
    __bf16* __restrict__ vt,   // [s][h][32][k]  (transposed)
    __bf16* __restrict__ gb)   // [row=65536][128]  sigmoid applied
{
    const int rt   = blockIdx.x;
    const int pj   = blockIdx.y;
    const int wave = threadIdx.x >> 5;
    const int lane = threadIdx.x & 31;
    const int half = lane >> 4;
    const int lm   = lane & 15;
    const int R0   = rt * 16;
    const int o0   = wave * 16;

    const float* x = (pj == 0 || pj == 3) ? qx : kvx;
    const float* w = (pj == 0) ? wq : (pj == 1) ? wk : (pj == 2) ? wv : wg;

    v8f acc = zero_v8f();
#pragma unroll
    for (int c = 0; c < 4; ++c) {
        v16bf a = load_a_f32(x, 128, R0, c * 32, lm, half);
        v16bf b = load_bT_f32(w, 128, o0, c * 32, lm, half);
        acc = wmma_bf16(a, b, acc);
    }

    const int col = o0 + lm;            // output feature 0..127
    const int h   = col >> 5;
    const int d   = col & 31;
#pragma unroll
    for (int r = 0; r < 8; ++r) {
        const int row  = R0 + r + 8 * half;   // C layout: M = r + 8*(lane/16)
        const int s    = row >> 8;
        const int qpos = row & 255;
        float v = acc[r];
        if (pj == 0) {
            v *= 0.17677669529663687f;  // 1/sqrt(32)
            qb[((size_t)((s * 4 + h) * 256 + qpos) << 5) + d] = (__bf16)v;
        } else if (pj == 1) {
            kb[((size_t)((s * 4 + h) * 256 + qpos) << 5) + d] = (__bf16)v;
        } else if (pj == 2) {
            vt[(size_t)((s * 4 + h) * 32 + d) * 256 + qpos] = (__bf16)v;
        } else {
            float g = v + bg[col];
            g = 1.0f / (1.0f + __expf(-g));
            gb[(size_t)row * 128 + col] = (__bf16)g;
        }
    }
}

// ---------------------------------------------------------------------------
// Kernel 2: attention per (s, h, q-tile).  One wave per block.
//   scores(16x256) = Q(16x32) @ K^T  (16 WMMAs, K-dim = D = 32)
//   + bias1[h,q,k] + bias2[s,k]; softmax over k; probs @ V (16 WMMAs);
//   fused sigmoid gate; bf16 output row-major [row][128].
// ---------------------------------------------------------------------------
__global__ __launch_bounds__(32) void attn_kernel(
    const __bf16* __restrict__ qb, const __bf16* __restrict__ kb,
    const __bf16* __restrict__ vt,
    const float* __restrict__ bias1,   // [4][256][256]
    const float* __restrict__ bias2,   // [256][256]
    const __bf16* __restrict__ gb,     // [65536][128]
    __bf16* __restrict__ ob)           // [65536][128]
{
    const int qt = blockIdx.x;   // 0..15
    const int h  = blockIdx.y;   // 0..3
    const int s  = blockIdx.z;   // 0..255
    const int lane = threadIdx.x & 31;
    const int half = lane >> 4;
    const int lm   = lane & 15;
    const int q0   = qt * 16;

    __shared__ __bf16 probs[16 * 256];   // 8 KB

    const __bf16* qbase = qb + ((size_t)(s * 4 + h) * 256) * 32;
    const __bf16* kbase = kb + ((size_t)(s * 4 + h) * 256) * 32;
    const __bf16* vbase = vt + ((size_t)(s * 4 + h) * 32) * 256;
    const float*  b1    = bias1 + ((size_t)h * 256 + q0) * 256;  // [qm][k]
    const float*  b2    = bias2 + (size_t)s * 256;

    // Q A-fragment (row-major bf16, ld = 32)
    v16bf aq = load_a_bf16(qbase, 32, q0, 0, lm, half);

    // ---- scores: 16 tiles of 16 columns, kept in registers (C layout) ----
    float sc[16][8];
#pragma unroll
    for (int t = 0; t < 16; ++t) {
        const int k0 = t * 16;
        // B column n -> K-row (k0+n), 32 contiguous d values
        v16bf bk = load_b_contig_bf16(kbase + (size_t)(k0 + lm) * 32, half);
        v8f accs = wmma_bf16(aq, bk, zero_v8f());
        const int   kcol = k0 + lm;
        const float bb2  = b2[kcol];
#pragma unroll
        for (int r = 0; r < 8; ++r) {
            const int qm = r + 8 * half;
            sc[t][r] = accs[r] + b1[qm * 256 + kcol] + bb2;
        }
    }

    // ---- softmax over k (row lives in one 16-lane half + 16 reg tiles) ----
#pragma unroll
    for (int r = 0; r < 8; ++r) {
        float mx = sc[0][r];
#pragma unroll
        for (int t = 1; t < 16; ++t) mx = fmaxf(mx, sc[t][r]);
#pragma unroll
        for (int dlt = 1; dlt < 16; dlt <<= 1)
            mx = fmaxf(mx, __shfl_xor(mx, dlt, 32));
        float sum = 0.0f;
#pragma unroll
        for (int t = 0; t < 16; ++t) {
            float e = __expf(sc[t][r] - mx);
            sc[t][r] = e;
            sum += e;
        }
#pragma unroll
        for (int dlt = 1; dlt < 16; dlt <<= 1)
            sum += __shfl_xor(sum, dlt, 32);
        const float inv = 1.0f / sum;
#pragma unroll
        for (int t = 0; t < 16; ++t) sc[t][r] *= inv;
    }

    // ---- C-layout -> A-layout re-distribution through LDS ----
#pragma unroll
    for (int t = 0; t < 16; ++t) {
        const int kcol = t * 16 + lm;
#pragma unroll
        for (int r = 0; r < 8; ++r)
            probs[(r + 8 * half) * 256 + kcol] = (__bf16)sc[t][r];
    }
    __syncthreads();

    // ---- o(16x32) = probs(16x256) @ V(256x32): 8 K-chunks x 2 N-tiles ----
    v8f o0acc = zero_v8f();
    v8f o1acc = zero_v8f();
#pragma unroll
    for (int c = 0; c < 8; ++c) {
        v16bf ap = load_a_bf16(probs, 256, 0, c * 32, lm, half);
        v16bf bv0 = load_b_contig_bf16(vbase + (size_t)lm * 256 + c * 32, half);
        v16bf bv1 = load_b_contig_bf16(vbase + (size_t)(16 + lm) * 256 + c * 32, half);
        o0acc = wmma_bf16(ap, bv0, o0acc);
        o1acc = wmma_bf16(ap, bv1, o1acc);
    }

    // ---- fused sigmoid gate, store bf16 [row][h*32+d] ----
#pragma unroll
    for (int r = 0; r < 8; ++r) {
        const int q   = q0 + r + 8 * half;
        const size_t rowoff = ((size_t)s * 256 + q) * 128;
        const int c0 = h * 32 + lm;
        const int c1 = h * 32 + 16 + lm;
        float g0 = (float)gb[rowoff + c0];
        float g1 = (float)gb[rowoff + c1];
        ob[rowoff + c0] = (__bf16)(o0acc[r] * g0);
        ob[rowoff + c1] = (__bf16)(o1acc[r] * g1);
    }
}

// ---------------------------------------------------------------------------
// Kernel 3: output projection  out = (o*g) @ wo^T + bo   (f32 result)
// ---------------------------------------------------------------------------
__global__ __launch_bounds__(256) void out_kernel(
    const __bf16* __restrict__ ob, const float* __restrict__ wo,
    const float* __restrict__ bo, float* __restrict__ out)
{
    const int rt   = blockIdx.x;
    const int wave = threadIdx.x >> 5;
    const int lane = threadIdx.x & 31;
    const int half = lane >> 4;
    const int lm   = lane & 15;
    const int R0   = rt * 16;
    const int o0   = wave * 16;

    v8f acc = zero_v8f();
#pragma unroll
    for (int c = 0; c < 4; ++c) {
        v16bf a = load_a_bf16(ob, 128, R0, c * 32, lm, half);
        v16bf b = load_bT_f32(wo, 128, o0, c * 32, lm, half);
        acc = wmma_bf16(a, b, acc);
    }

    const int col = o0 + lm;
    const float bias = bo[col];
#pragma unroll
    for (int r = 0; r < 8; ++r) {
        const int row = R0 + r + 8 * half;
        out[(size_t)row * 128 + col] = acc[r] + bias;
    }
}

// ---------------------------------------------------------------------------
// Host launcher
// ---------------------------------------------------------------------------
extern "C" void kernel_launch(void* const* d_in, const int* in_sizes, int n_in,
                              void* d_out, int out_size, void* d_ws, size_t ws_size,
                              hipStream_t stream) {
    const float* q_x   = (const float*)d_in[0];   // (1,256,256,128)
    const float* kv_x  = (const float*)d_in[1];   // (1,256,256,128)
    const float* bias1 = (const float*)d_in[2];   // (1,1,4,256,256)
    const float* bias2 = (const float*)d_in[3];   // (1,256,1,1,256)
    const float* wq    = (const float*)d_in[4];
    const float* wk    = (const float*)d_in[5];
    const float* wv    = (const float*)d_in[6];
    const float* wg    = (const float*)d_in[7];
    const float* bg    = (const float*)d_in[8];
    const float* wo    = (const float*)d_in[9];
    const float* bo    = (const float*)d_in[10];
    float*       out   = (float*)d_out;

    // Workspace carve-out (bf16 intermediates), 5 x 16 MiB = 80 MiB
    char* ws = (char*)d_ws;
    const size_t SZ = (size_t)256 * 4 * 256 * 32 * sizeof(__bf16); // 16 MiB
    __bf16* qb = (__bf16*)(ws + 0 * SZ);   // [s][h][q][32]
    __bf16* kb = (__bf16*)(ws + 1 * SZ);   // [s][h][k][32]
    __bf16* vt = (__bf16*)(ws + 2 * SZ);   // [s][h][32][k]
    __bf16* gb = (__bf16*)(ws + 3 * SZ);   // [65536][128]
    __bf16* ob = (__bf16*)(ws + 4 * SZ);   // [65536][128]

    // 1) fused projections (Q scaled, V transposed, G sigmoid'ed)
    proj_kernel<<<dim3(4096, 4), 256, 0, stream>>>(q_x, kv_x, wq, wk, wv, wg,
                                                   bg, qb, kb, vt, gb);
    // 2) attention + gate
    attn_kernel<<<dim3(16, 4, 256), 32, 0, stream>>>(qb, kb, vt, bias1, bias2,
                                                     gb, ob);
    // 3) output projection
    out_kernel<<<dim3(4096), 256, 0, stream>>>(ob, wo, bo, out);
}
